// FullyGRUNetwork_61452392071409
// MI455X (gfx1250) — compile-verified, matrix operand order checked by reading, not run on hardware
//
#include <hip/hip_runtime.h>
#include <hip/hip_bf16.h>
#include <cstdint>

// ---------------------------------------------------------------------------
// Shapes
// ---------------------------------------------------------------------------
#define Bsz  512
#define Tt   1024
#define Ff   128
#define Hh   81
#define Hp   96          // padded hidden (6 WMMA n-tiles of 16)
#define Gp   288         // 3 * Hp padded gate rows (r | z | n)
#define NBT  32          // batch tiles of 16 rows
#define NGT  18          // gate n-tiles (3 gates x 6)

typedef __attribute__((ext_vector_type(16))) __bf16 v16bf;
typedef __attribute__((ext_vector_type(8)))  __bf16 v8bf;
typedef __attribute__((ext_vector_type(8)))  float  v8f;
typedef __attribute__((ext_vector_type(4)))  unsigned int v4u;
typedef __attribute__((ext_vector_type(8)))  int    v8i;
typedef __attribute__((ext_vector_type(4)))  int    v4i;

#if defined(__has_builtin)
#if __has_builtin(__builtin_amdgcn_tensor_load_to_lds) && \
    __has_builtin(__builtin_amdgcn_s_wait_tensorcnt)
#define HAVE_TDM 1
#endif
#endif

// ---------------------------------------------------------------------------
// WMMA helper: D = A(16x32 bf16) x B(32x16 bf16) + C(16x16 f32)
// ---------------------------------------------------------------------------
__device__ __forceinline__ v8f wmma_bf16(v16bf a, v16bf b, v8f c) {
    return __builtin_amdgcn_wmma_f32_16x16x32_bf16(
        /*neg_a=*/false, a, /*neg_b=*/false, b,
        /*c_mod=*/(short)0, c, /*reuse_a=*/false, /*reuse_b=*/false);
}

// A/B tile operand for lane L from bf16 memory: two contiguous 16-byte runs
// (ISA 16-bit 16x32 layout; caller pre-offsets by half*8 elements).
__device__ __forceinline__ v16bf load_tile_runs(const __bf16* p) {
    v8bf lo = *(const v8bf*)(p);
    v8bf hi = *(const v8bf*)(p + 16);
    v16bf t;
#pragma unroll
    for (int i = 0; i < 8; ++i) { t[i] = lo[i]; t[i + 8] = hi[i]; }
    return t;
}

// Same tile operand built from f32 memory (converted to bf16 in registers).
__device__ __forceinline__ v16bf load_tile_runs_f32(const float* p) {
    float4 a0 = *(const float4*)(p);
    float4 a1 = *(const float4*)(p + 4);
    float4 b0 = *(const float4*)(p + 16);
    float4 b1 = *(const float4*)(p + 20);
    v16bf t;
    t[0]  = (__bf16)a0.x; t[1]  = (__bf16)a0.y; t[2]  = (__bf16)a0.z; t[3]  = (__bf16)a0.w;
    t[4]  = (__bf16)a1.x; t[5]  = (__bf16)a1.y; t[6]  = (__bf16)a1.z; t[7]  = (__bf16)a1.w;
    t[8]  = (__bf16)b0.x; t[9]  = (__bf16)b0.y; t[10] = (__bf16)b0.z; t[11] = (__bf16)b0.w;
    t[12] = (__bf16)b1.x; t[13] = (__bf16)b1.y; t[14] = (__bf16)b1.z; t[15] = (__bf16)b1.w;
    return t;
}

__device__ __forceinline__ v8f load8f(const float* p) {
    float4 a = *(const float4*)(p);
    float4 b = *(const float4*)(p + 4);
    v8f r;
    r[0] = a.x; r[1] = a.y; r[2] = a.z; r[3] = a.w;
    r[4] = b.x; r[5] = b.y; r[6] = b.z; r[7] = b.w;
    return r;
}

__device__ __forceinline__ void store8f(float* p, v8f v) {
    *(float4*)(p)     = make_float4(v[0], v[1], v[2], v[3]);
    *(float4*)(p + 4) = make_float4(v[4], v[5], v[6], v[7]);
}

__device__ __forceinline__ v8f splat8(float v) {
    v8f r;
#pragma unroll
    for (int i = 0; i < 8; ++i) r[i] = v;
    return r;
}

__device__ __forceinline__ float sigmoidf_(float x) {
    return 1.0f / (1.0f + __expf(-x));
}

__device__ __forceinline__ float tanhf_fast(float v) {
    float e = __expf(-2.0f * v);          // one v_exp on the critical path
    return (1.0f - e) / (1.0f + e);
}

// ---------------------------------------------------------------------------
// TDM: DMA a contiguous run of `nelem` bf16 elements from global to LDS.
// D# per CDNA5 ISA §8.3/8.4: count=1, type=2, data_size=1 (2 bytes),
// 1-D tile (tile_dim0 = nelem < 65536), groups 2/3 zero (<=2D tensor).
// This toolchain exposes the 6-arg builtin (extra int32x8 descriptor group).
// Issued by one wave; caller barriers afterwards.
// ---------------------------------------------------------------------------
#ifdef HAVE_TDM
__device__ __forceinline__ void tdm_load_bf16(void* ldsDst, const void* gsrc,
                                              unsigned nelem) {
    unsigned long long ga = (unsigned long long)(uintptr_t)gsrc;
    unsigned ldsOff = (unsigned)(uintptr_t)ldsDst;   // LDS aperture low bits

    v4u g0;
    g0[0] = 1u;                                      // count=1, user mode
    g0[1] = ldsOff;                                  // lds_addr
    g0[2] = (unsigned)(ga & 0xffffffffu);            // global_addr[31:0]
    g0[3] = (unsigned)((ga >> 32) & 0x01ffffffu)     // global_addr[56:32]
          | (2u << 30);                              // type=2 ("image")

    v8i g1;
    g1[0] = (int)(1u << 16);                         // wg_mask=0, data_size=2B
    g1[1] = (int)((nelem & 0xffffu) << 16);          // tensor_dim0[15:0]
    g1[2] = (int)(((nelem >> 16) & 0xffffu)          // tensor_dim0[31:16]
          | (1u << 16));                             // tensor_dim1 = 1
    g1[3] = (int)((nelem & 0xffffu) << 16);          // tile_dim0 = nelem
    g1[4] = 0;                                       // tile_dim1/2 unused
    g1[5] = (int)nelem;                              // tensor_dim0_stride lo
    g1[6] = 0;                                       // stride hi / dim1_stride
    g1[7] = 0;

    v4i g2 = {0, 0, 0, 0};
    v4i g3 = {0, 0, 0, 0};
    v8i g4 = {0, 0, 0, 0, 0, 0, 0, 0};
    __builtin_amdgcn_tensor_load_to_lds(g0, g1, g2, g3, g4, 0);
    __builtin_amdgcn_s_wait_tensorcnt(0);
}
#endif

// ---------------------------------------------------------------------------
// Kernel 1: build zero-padded bf16 weights + f32 biases.
// Padded gate row g' = gate*96 + j (gate 0=r,1=z,2=n; zero for j>=81)
// ---------------------------------------------------------------------------
__global__ void prep_kernel(const float* __restrict__ W_ih, const float* __restrict__ W_hh,
                            const float* __restrict__ b_ih, const float* __restrict__ b_hh,
                            __bf16* __restrict__ wp_ih, __bf16* __restrict__ wp_hh,
                            float* __restrict__ bp_ih, float* __restrict__ bp_hh) {
    int tid = blockIdx.x * blockDim.x + threadIdx.x;
    int nthr = gridDim.x * blockDim.x;
    for (int idx = tid; idx < Gp * Ff; idx += nthr) {
        int gp = idx / Ff, k = idx % Ff;
        int gate = gp / Hp, j = gp % Hp;
        float v = (j < Hh) ? W_ih[(gate * Hh + j) * Ff + k] : 0.0f;
        wp_ih[idx] = (__bf16)v;
    }
    for (int idx = tid; idx < Gp * Hp; idx += nthr) {
        int gp = idx / Hp, k = idx % Hp;
        int gate = gp / Hp, j = gp % Hp;
        float v = (j < Hh && k < Hh) ? W_hh[(gate * Hh + j) * Hh + k] : 0.0f;
        wp_hh[idx] = (__bf16)v;
    }
    for (int idx = tid; idx < Gp; idx += nthr) {
        int gate = idx / Hp, j = idx % Hp;
        bp_ih[idx] = (j < Hh) ? b_ih[gate * Hh + j] : 0.0f;
        bp_hh[idx] = (j < Hh) ? b_hh[gate * Hh + j] : 0.0f;
    }
}

// ---------------------------------------------------------------------------
// Kernel 2: full-occupancy input projection GEMM.
//   xg[t, btile, gt, lane, 0..7] = (x_t @ W_ih^T + b_ih) tile in WMMA C-layout
// One wave per (t, btile); 18 gate tiles x 4 K-tiles = 72 WMMAs per wave.
// ---------------------------------------------------------------------------
__global__ void __launch_bounds__(256) xg_gemm_kernel(
    const float* __restrict__ x,
    const __bf16* __restrict__ wp_ih, const float* __restrict__ bp_ih,
    float* __restrict__ xg) {

    extern __shared__ __align__(16) char smem[];
    __bf16* sWih = (__bf16*)smem;              // Gp * Ff
    float*  sBih = (float*)(sWih + Gp * Ff);   // Gp

    const int tid  = threadIdx.x;
    const int lane = tid & 31;
    const int wave = tid >> 5;
    const int m    = lane & 15;
    const int hf   = lane >> 4;

#ifdef HAVE_TDM
    if (tid < 32) tdm_load_bf16(sWih, wp_ih, Gp * Ff);   // wave 0 DMA + wait
#else
    for (int i = tid; i < Gp * Ff; i += 256) sWih[i] = wp_ih[i];
#endif
    for (int i = tid; i < Gp; i += 256) sBih[i] = bp_ih[i];
    __syncthreads();

    const int wgid  = blockIdx.x * 8 + wave;   // 0 .. 32767
    const int t     = wgid >> 5;               // 0 .. 1023
    const int btile = wgid & 31;               // 0 .. 31
    const int b0    = btile * 16;

    // A operand: 4 K-tiles of x_t rows (fp32 -> bf16 in regs)
    const float* xp = x + (size_t)(b0 + m) * (size_t)(Tt * Ff) + (size_t)t * Ff;
    v16bf Ax[4];
#pragma unroll
    for (int kt = 0; kt < 4; ++kt)
        Ax[kt] = load_tile_runs_f32(xp + kt * 32 + hf * 8);

    float* outb = xg + ((((size_t)t * NBT + btile) * NGT) * 32 + lane) * 8;

#pragma unroll
    for (int gt = 0; gt < NGT; ++gt) {
        const int nr = gt * 16 + m;            // padded gate row
        v8f acc = splat8(sBih[nr]);            // fold b_ih here
#pragma unroll
        for (int kt = 0; kt < 4; ++kt) {
            const int ko = kt * 32 + hf * 8;
            acc = wmma_bf16(Ax[kt], load_tile_runs(sWih + (size_t)nr * Ff + ko), acc);
        }
        store8f(outb + (size_t)gt * 32 * 8, acc);
    }
}

// ---------------------------------------------------------------------------
// Kernel 3: recurrent loop. 32 blocks x 6 waves; wave w owns hidden n-tile w.
// Shared h state ping-pongs between two LDS buffers (1 barrier / step).
// Per step per wave: 6 global b128 (xg), 6 ds b128 (h), 9 WMMAs, gate math.
// ---------------------------------------------------------------------------
__global__ void __launch_bounds__(192) gru_rec_kernel(
    const float* __restrict__ xg,
    const __bf16* __restrict__ wp_hh, const float* __restrict__ bp_hh,
    float* __restrict__ hout) {

    extern __shared__ __align__(16) char smem[];
    __bf16* sWhh = (__bf16*)smem;                  // Gp * Hp
    float*  sBhh = (float*)(sWhh + Gp * Hp);       // Gp
    __bf16* sH0  = (__bf16*)(sBhh + Gp);           // 16 * Hp
    __bf16* sH1  = sH0 + 16 * Hp;                  // 16 * Hp

    const int tid  = threadIdx.x;
    const int lane = tid & 31;
    const int nt   = tid >> 5;      // wave id == hidden n-tile 0..5
    const int m    = lane & 15;
    const int hf   = lane >> 4;

#ifdef HAVE_TDM
    if (tid < 32) tdm_load_bf16(sWhh, wp_hh, Gp * Hp);   // wave 0 DMA + wait
#else
    for (int i = tid; i < Gp * Hp; i += 192) sWhh[i] = wp_hh[i];
#endif
    for (int i = tid; i < Gp; i += 192) sBhh[i] = bp_hh[i];
    for (int i = tid; i < 2 * 16 * Hp; i += 192) sH0[i] = (__bf16)0.0f;
    __syncthreads();

    const int btile = blockIdx.x;
    const int b0    = btile * 16;
    const int nr    = nt * 16 + m;
    const float bhr = sBhh[nr];
    const float bhz = sBhh[nr + Hp];
    const float bhn = sBhh[nr + 2 * Hp];

    v8f h = splat8(0.0f);

    // xg tile pointers for this wave: gate tiles nt (r), nt+6 (z), nt+12 (n)
    const size_t stepStride = (size_t)NBT * NGT * 32 * 8;
    const float* xr_p = xg + (((size_t)btile * NGT + nt) * 32 + lane) * 8;
    const float* xz_p = xr_p + (size_t)6 * 32 * 8;
    const float* xn_p = xr_p + (size_t)12 * 32 * 8;

#pragma unroll 1
    for (int t = 0; t < Tt; ++t) {
        const size_t so = (size_t)t * stepStride;
        v8f xr = load8f(xr_p + so);
        v8f xz = load8f(xz_p + so);
        v8f xn = load8f(xn_p + so);
        if (t + 1 < Tt) {   // CDNA5 global_prefetch of next step's tiles
            __builtin_prefetch(xr_p + so + stepStride, 0, 1);
            __builtin_prefetch(xn_p + so + stepStride, 0, 1);
        }

        const __bf16* rbuf = (t & 1) ? sH1 : sH0;
        __bf16*       wbuf = (t & 1) ? sH0 : sH1;

        v16bf Ah[3];
#pragma unroll
        for (int kt = 0; kt < 3; ++kt)
            Ah[kt] = load_tile_runs(rbuf + m * Hp + kt * 32 + hf * 8);

        v8f hr = splat8(bhr), hz = splat8(bhz), hn = splat8(bhn);
#pragma unroll
        for (int kt = 0; kt < 3; ++kt) {
            const int ko = kt * 32 + hf * 8;
            hr = wmma_bf16(Ah[kt], load_tile_runs(sWhh + (size_t)nr * Hp + ko), hr);
            hz = wmma_bf16(Ah[kt], load_tile_runs(sWhh + (size_t)(nr + Hp) * Hp + ko), hz);
            hn = wmma_bf16(Ah[kt], load_tile_runs(sWhh + (size_t)(nr + 2 * Hp) * Hp + ko), hn);
        }

        v8f hnew;
#pragma unroll
        for (int i = 0; i < 8; ++i) {
            float r = sigmoidf_(xr[i] + hr[i]);
            float z = sigmoidf_(xz[i] + hz[i]);
            float n = tanhf_fast(xn[i] + r * hn[i]);
            hnew[i] = (1.0f - z) * n + z * h[i];
        }
        h = hnew;

#pragma unroll
        for (int i = 0; i < 8; ++i) {
            const int mm = i + hf * 8;                   // D-layout row
            wbuf[mm * Hp + nt * 16 + m] = (__bf16)hnew[i];
        }
        __syncthreads();  // writes of step t visible; reads of step t done
    }

#pragma unroll
    for (int i = 0; i < 8; ++i) {
        const int mm = i + hf * 8;
        hout[(size_t)(b0 + mm) * Hp + nt * 16 + m] = h[i];
    }
}

// ---------------------------------------------------------------------------
// Kernel 4: head. fc(81->8)+silu, BN, fc(8->4)+silu, BN, fc(4->1)
// ---------------------------------------------------------------------------
__global__ void __launch_bounds__(512) head_kernel(
    const float* __restrict__ hbuf,      // [Bsz, Hp]
    const float* __restrict__ fc_w, const float* __restrict__ fc_b,
    const float* __restrict__ g1, const float* __restrict__ be1,
    const float* __restrict__ fc1w, const float* __restrict__ fc1b,
    const float* __restrict__ g2, const float* __restrict__ be2,
    const float* __restrict__ fc2w, const float* __restrict__ fc2b,
    float* __restrict__ out) {

    __shared__ float sv[Bsz * 8];
    __shared__ float stats[16];          // [0..7]=mu, [8..15]=rsig

    const int b = threadIdx.x;
    const float* h = hbuf + (size_t)b * Hp;

    float o8[8];
#pragma unroll
    for (int c = 0; c < 8; ++c) {
        float acc = fc_b[c];
        for (int j = 0; j < Hh; ++j) acc += h[j] * fc_w[c * Hh + j];
        float s = acc * sigmoidf_(acc);  // silu
        o8[c] = s;
        sv[b * 8 + c] = s;
    }
    __syncthreads();
    if (b < 8) {
        float mu = 0.0f, m2 = 0.0f;
        for (int i = 0; i < Bsz; ++i) { float v = sv[i * 8 + b]; mu += v; m2 += v * v; }
        mu *= (1.0f / Bsz);
        m2 = m2 * (1.0f / Bsz) - mu * mu;
        stats[b] = mu;
        stats[8 + b] = rsqrtf(m2 + 1e-5f);
    }
    __syncthreads();

    float p4[4];
#pragma unroll
    for (int c2 = 0; c2 < 4; ++c2) {
        float acc = fc1b[c2];
#pragma unroll
        for (int c = 0; c < 8; ++c) {
            float v = (o8[c] - stats[c]) * stats[8 + c] * g1[c] + be1[c];
            acc += v * fc1w[c2 * 8 + c];
        }
        float s = acc * sigmoidf_(acc);  // silu
        p4[c2] = s;
        sv[b * 8 + c2] = s;
    }
    __syncthreads();
    if (b < 4) {
        float mu = 0.0f, m2 = 0.0f;
        for (int i = 0; i < Bsz; ++i) { float v = sv[i * 8 + b]; mu += v; m2 += v * v; }
        mu *= (1.0f / Bsz);
        m2 = m2 * (1.0f / Bsz) - mu * mu;
        stats[b] = mu;
        stats[8 + b] = rsqrtf(m2 + 1e-5f);
    }
    __syncthreads();

    float acc = fc2b[0];
#pragma unroll
    for (int c2 = 0; c2 < 4; ++c2) {
        float v = (p4[c2] - stats[c2]) * stats[8 + c2] * g2[c2] + be2[c2];
        acc += v * fc2w[c2];
    }
    out[b] = acc;
}

// ---------------------------------------------------------------------------
// Launcher
// ---------------------------------------------------------------------------
extern "C" void kernel_launch(void* const* d_in, const int* in_sizes, int n_in,
                              void* d_out, int out_size, void* d_ws, size_t ws_size,
                              hipStream_t stream) {
    const float* x    = (const float*)d_in[0];
    const float* W_ih = (const float*)d_in[1];
    const float* W_hh = (const float*)d_in[2];
    const float* b_ih = (const float*)d_in[3];
    const float* b_hh = (const float*)d_in[4];
    const float* fc_w = (const float*)d_in[5];
    const float* fc_b = (const float*)d_in[6];
    const float* g1   = (const float*)d_in[7];
    const float* be1  = (const float*)d_in[8];
    const float* fc1w = (const float*)d_in[9];
    const float* fc1b = (const float*)d_in[10];
    const float* g2   = (const float*)d_in[11];
    const float* be2  = (const float*)d_in[12];
    const float* fc2w = (const float*)d_in[13];
    const float* fc2b = (const float*)d_in[14];
    float* out = (float*)d_out;

    char* ws = (char*)d_ws;
    size_t off = 0;
    auto carve = [&](size_t bytes) -> char* {
        char* p = ws + off;
        off = (off + bytes + 255) & ~(size_t)255;
        return p;
    };
    float*  xgbuf = (float*)carve((size_t)Tt * NBT * NGT * 32 * 8 * 4); // 604 MB
    __bf16* wp_ih = (__bf16*)carve((size_t)Gp * Ff * 2);
    __bf16* wp_hh = (__bf16*)carve((size_t)Gp * Hp * 2);
    float*  bp_ih = (float*)carve((size_t)Gp * 4);
    float*  bp_hh = (float*)carve((size_t)Gp * 4);
    float*  hbuf  = (float*)carve((size_t)Bsz * Hp * 4);

    // 1) padded weights/biases
    prep_kernel<<<64, 256, 0, stream>>>(W_ih, W_hh, b_ih, b_hh,
                                        wp_ih, wp_hh, bp_ih, bp_hh);

    // 2) input projection at full occupancy: 4096 blocks x 8 waves
    size_t smem_g = (size_t)Gp * Ff * 2 + (size_t)Gp * 4;
    xg_gemm_kernel<<<4096, 256, smem_g, stream>>>(x, wp_ih, bp_ih, xgbuf);

    // 3) recurrence: 32 blocks x 6 cooperating waves
    size_t smem_r = (size_t)Gp * Hp * 2 + (size_t)Gp * 4
                  + (size_t)2 * 16 * Hp * 2;
    gru_rec_kernel<<<NBT, 192, smem_r, stream>>>(xgbuf, wp_hh, bp_hh, hbuf);

    // 4) head
    head_kernel<<<1, 512, 0, stream>>>(hbuf, fc_w, fc_b, g1, be1,
                                       fc1w, fc1b, g2, be2, fc2w, fc2b, out);
}
